// GNNDecoder_1632087573108
// MI455X (gfx1250) — matrix operand order repaired and checked by reading, not compiled, not run on hardware
//
#include <hip/hip_runtime.h>

// ---------------------------------------------------------------------------
// GNN decoder for MI455X (gfx1250, wave32).
// GEMMs: v_wmma_f32_16x16x32_bf16, B-tiles staged to LDS with
// global_load_async_to_lds_b128 (ASYNCcnt, double-buffered), A streamed from
// global with global_prefetch_b8. Weights pre-transposed to bf16 [N][K]
// (padded to 128-col multiples so the K-loop has no EXEC juggling).
// Edge layer chunked (16 x 4096 edges) to bound workspace.
// ---------------------------------------------------------------------------

using bf16 = __bf16;
typedef __attribute__((ext_vector_type(16))) __bf16 bf16x16;
typedef __attribute__((ext_vector_type(8)))  float  floatx8;

#define NN    4096
#define EE    65536
#define DIN   2048
#define DD    512
#define HH    8
#define DHID  2048
#define NCLS  98
#define NCLSP 128    // cls cols padded to col-tile multiple
#define CEC   4096   // edge chunk size

__device__ __forceinline__ float gelu_f(float x) {
  return 0.5f * x * (1.0f + erff(x * 0.70710678118654752f));
}
__device__ __forceinline__ float sigmoid_f(float x) {
  return 1.0f / (1.0f + expf(-x));
}

union Frag { uint4 u[2]; bf16x16 v; };

// ---------------------------------------------------------------------------
// WMMA GEMM: C[M x Ncol] = act(A[M x K] @ W[K x Ncol] + bias)
//   A    : bf16 row-major, optionally row-gathered through aidx[]
//   BT   : bf16 [NcolPad][K] (W transposed; rows padded/zeroed to 128-mult)
//   sidx : if non-null, atomicAdd rows into Cf[sidx[row]] (segment scatter)
//   act  : 0 none, 1 gelu, 2 sigmoid
// Block = 256 threads = 8 waves = 128 rows x 128 cols. Wave = 16 x 128
// (8 accumulators). B tile (128 x 32K = 8 KB) double-buffered in LDS, filled
// with global_load_async_to_lds_b128. REQUIRES M % 128 == 0.
// ---------------------------------------------------------------------------
__global__ __launch_bounds__(256)
void k_gemm(const bf16* __restrict__ A, const bf16* __restrict__ BT,
            const float* __restrict__ bias,
            const int* __restrict__ aidx, const int* __restrict__ sidx,
            float* Cf, bf16* Cb,
            int M, int Ncol, int K, int act)
{
  const int wid  = threadIdx.x >> 5;
  const int lane = threadIdx.x & 31;
  const int l16  = lane & 15;
  const int half = lane >> 4;
  const int row0 = blockIdx.x * 128 + wid * 16;
  const int col0 = blockIdx.y * 128;

  const int m = row0 + l16;
  const size_t arow = aidx ? (size_t)aidx[m] : (size_t)m;
  const bf16* Ap = A + arow * (size_t)K;
  const bf16* Bbase = BT + (size_t)col0 * K;

  // LDS B tile: 128 rows (cols of C) x 64 bytes (32 K-values, bf16), x2 bufs.
  __shared__ __align__(16) char sB[2][128 * 64];

  // Async-fill one buffer with the 128x32 B tile at K-offset kk.
  // 512 units of 16 bytes, 256 threads -> 2 async b128 copies per thread.
  auto stage = [&](int bi, int kk) {
#pragma unroll
    for (int i = 0; i < 2; ++i) {
      const int u    = threadIdx.x + i * 256;
      const int n    = u >> 2;
      const int part = u & 3;
      const bf16* gp = Bbase + (size_t)n * K + kk + part * 8;
      const unsigned lo = (unsigned)(size_t)(&sB[bi][n * 64 + part * 16]);
      asm volatile("global_load_async_to_lds_b128 %0, %1, off"
                   :: "v"(lo), "v"(gp) : "memory");
    }
  };

  floatx8 acc[8] = {};

  int buf = 0;
  stage(0, 0);
  for (int kk = 0; kk < K; kk += 32) {
    asm volatile("s_wait_asynccnt 0" ::: "memory");
    __syncthreads();
    if (kk + 32 < K) stage(buf ^ 1, kk + 32);

    // A fragment, 16-bit A-matrix 16x32 layout (ISA 7.12.2):
    // elem j<8 : K = kk + half*8 + j ; elem j>=8 : K = kk + 16 + half*8 + (j-8)
    Frag a;
    a.u[0] = *(const uint4*)(Ap + kk + half * 8);
    a.u[1] = *(const uint4*)(Ap + kk + 16 + half * 8);
    __builtin_prefetch(Ap + kk + 128, 0, 3);   // global_prefetch_b8, near

#pragma unroll
    for (int j = 0; j < 8; ++j) {
      // B fragment: lane covers K = kk + half*8 .. +7 and kk + 16 + half*8 .. +7 of column col0+j*16+l16
      const char* lp = &sB[buf][(j * 16 + l16) * 64 + half * 16];
      Frag b;
      b.u[0] = ((const uint4*)lp)[0];
      b.u[1] = ((const uint4*)(lp + 32))[0];
      acc[j] = __builtin_amdgcn_wmma_f32_16x16x32_bf16(
          false, a.v, false, b.v, (short)0, acc[j], false, false);
    }
    buf ^= 1;
  }

  // Epilogue. C/D layout: VGPR p, lane -> row = row0 + p + half*8, col = col0+j*16+l16
#pragma unroll
  for (int j = 0; j < 8; ++j) {
    const int n = col0 + j * 16 + l16;
    if (n >= Ncol) continue;
    const float bv = bias ? bias[n] : 0.0f;
#pragma unroll
    for (int p = 0; p < 8; ++p) {
      const int grow = row0 + p + half * 8;
      float v = acc[j][p] + bv;
      if (act == 1) v = gelu_f(v);
      else if (act == 2) v = sigmoid_f(v);
      if (sidx) {
        atomicAdd(&Cf[(size_t)sidx[grow] * Ncol + n], v);
      } else {
        const size_t o = (size_t)grow * Ncol + n;
        if (Cf) Cf[o] = v;
        if (Cb) Cb[o] = (bf16)v;
      }
    }
  }
}

// --------------------------- small utility kernels -------------------------
__global__ void k_split(const int* __restrict__ ei, int* rA, int* cA) {
  int e = blockIdx.x * 256 + threadIdx.x;
  if (e < EE) { rA[e] = ei[2 * e]; cA[e] = ei[2 * e + 1]; }
}

__global__ void k_cvt(const float* __restrict__ x, bf16* __restrict__ y, size_t n) {
  size_t i = (size_t)blockIdx.x * 256 + threadIdx.x;
  if (i < n) y[i] = (bf16)x[i];
}

__global__ void k_wT(const float* __restrict__ W, bf16* __restrict__ WT, int K, int Ncol) {
  size_t idx = (size_t)blockIdx.x * 256 + threadIdx.x;
  if (idx >= (size_t)K * Ncol) return;
  int k = (int)(idx / Ncol), n = (int)(idx % Ncol);
  WT[(size_t)n * K + k] = (bf16)W[idx];
}

__global__ void k_zero_bf16(bf16* __restrict__ p, size_t n) {
  size_t i = (size_t)blockIdx.x * 256 + threadIdx.x;
  if (i < n) p[i] = (bf16)0.0f;
}

__global__ void k_kvidx(const int* __restrict__ rA, const int* __restrict__ cA,
                        int* __restrict__ kv, int n) {
  int i = blockIdx.x * 256 + threadIdx.x;
  if (i < n) { kv[2 * i] = rA[i]; kv[2 * i + 1] = cA[i]; }
}

// ------------------------ node attention (segment softmax) -----------------
__global__ void k_sdot(const bf16* __restrict__ qB, const bf16* __restrict__ kB,
                       const int* __restrict__ rA, const int* __restrict__ cA,
                       float* __restrict__ s) {
  size_t idx = (size_t)blockIdx.x * 256 + threadIdx.x;
  if (idx >= (size_t)EE * HH) return;
  int e = (int)(idx >> 3), h = (int)(idx & 7);
  const bf16* qp = qB + (size_t)cA[e] * DD + h * 64;
  const bf16* kp = kB + (size_t)rA[e] * DD + h * 64;
  float acc = 0.0f;
#pragma unroll 8
  for (int i = 0; i < 64; ++i) acc += (float)qp[i] * (float)kp[i];
  s[idx] = acc * 0.125f;   // 1/sqrt(64)
}

__global__ void k_segmax(const float* __restrict__ s, const int* __restrict__ cA,
                         unsigned* __restrict__ mEnc) {
  size_t idx = (size_t)blockIdx.x * 256 + threadIdx.x;
  if (idx >= (size_t)EE * HH) return;
  int e = (int)(idx >> 3), h = (int)(idx & 7);
  unsigned b = __float_as_uint(s[idx]);
  unsigned enc = (b & 0x80000000u) ? ~b : (b | 0x80000000u);  // order-preserving
  atomicMax(&mEnc[(size_t)cA[e] * HH + h], enc);
}

__global__ void k_mdec(const unsigned* __restrict__ mEnc, float* __restrict__ mDec) {
  size_t idx = (size_t)blockIdx.x * 256 + threadIdx.x;
  if (idx >= (size_t)NN * HH) return;
  unsigned enc = mEnc[idx];
  float m = 0.0f;
  if (enc != 0u) {
    unsigned b = (enc & 0x80000000u) ? (enc ^ 0x80000000u) : ~enc;
    m = __uint_as_float(b);
  }
  mDec[idx] = m;
}

__global__ void k_expz(const float* __restrict__ s, const float* __restrict__ mDec,
                       const int* __restrict__ cA, float* __restrict__ eExp,
                       float* __restrict__ z) {
  size_t idx = (size_t)blockIdx.x * 256 + threadIdx.x;
  if (idx >= (size_t)EE * HH) return;
  int e = (int)(idx >> 3), h = (int)(idx & 7);
  int c = cA[e];
  float ev = expf(s[idx] - mDec[(size_t)c * HH + h]);
  eExp[idx] = ev;
  atomicAdd(&z[(size_t)c * HH + h], ev);
}

__global__ void k_scatter_v(const float* __restrict__ eExp, const float* __restrict__ z,
                            const bf16* __restrict__ vB, const int* __restrict__ rA,
                            const int* __restrict__ cA, float* __restrict__ agg) {
  size_t idx = (size_t)blockIdx.x * 256 + threadIdx.x;
  if (idx >= (size_t)EE * DD) return;
  int e = (int)(idx >> 9);
  int d = (int)(idx & 511);
  int h = d >> 6;
  int r = rA[e], c = cA[e];
  float alpha = eExp[(size_t)e * HH + h] / z[(size_t)c * HH + h];
  atomicAdd(&agg[(size_t)c * DD + d], alpha * (float)vB[(size_t)r * DD + d]);
}

// ---------------- residual + gelu + LayerNorm (row-wise) -------------------
// out = LN(gelu(a + b)) * g + be ; R in {512, 2048}; in-place on b allowed.
__global__ __launch_bounds__(256)
void k_norm_rows(const float* a, const float* b,
                 const float* __restrict__ g, const float* __restrict__ be,
                 float* outF, bf16* outB, int R, int doGelu)
{
  const size_t base = (size_t)blockIdx.x * R;
  const int per = R >> 8;   // elems per thread (2 or 8)
  float vals[8];
  float s1 = 0.0f, s2 = 0.0f;
  for (int i = 0; i < per; ++i) {
    const int idx = threadIdx.x + (i << 8);
    float x = a[base + idx];
    if (b) x += b[base + idx];
    if (doGelu) x = gelu_f(x);
    vals[i] = x; s1 += x; s2 += x * x;
  }
#pragma unroll
  for (int o = 16; o > 0; o >>= 1) {
    s1 += __shfl_xor(s1, o, 32);
    s2 += __shfl_xor(s2, o, 32);
  }
  __shared__ float sh1[8], sh2[8];
  __shared__ float shmu, shinv;
  const int wid = threadIdx.x >> 5;
  if ((threadIdx.x & 31) == 0) { sh1[wid] = s1; sh2[wid] = s2; }
  __syncthreads();
  if (threadIdx.x == 0) {
    float t1 = 0.0f, t2 = 0.0f;
    for (int i = 0; i < 8; ++i) { t1 += sh1[i]; t2 += sh2[i]; }
    float mu = t1 / (float)R;
    float var = t2 / (float)R - mu * mu;
    shmu = mu; shinv = rsqrtf(var + 1e-5f);
  }
  __syncthreads();
  const float mu = shmu, inv = shinv;
  for (int i = 0; i < per; ++i) {
    const int idx = threadIdx.x + (i << 8);
    float y = (vals[i] - mu) * inv * g[idx] + be[idx];
    if (outF) outF[base + idx] = y;
    if (outB) outB[base + idx] = (bf16)y;
  }
}

// ---------------- edge 2-way attention (one wave per edge) -----------------
__global__ __launch_bounds__(256)
void k_edge_attn(const bf16* __restrict__ qB, const bf16* __restrict__ kB,
                 const bf16* __restrict__ vB, float* __restrict__ eatt, int n)
{
  const int wid  = threadIdx.x >> 5;
  const int lane = threadIdx.x & 31;
  const int e = blockIdx.x * 8 + wid;
  if (e >= n) return;
  const bf16* qp = qB + (size_t)e * DD;
  const bf16* k0 = kB + (size_t)(2 * e) * DD;
  const bf16* k1 = k0 + DD;
  float a0 = 0.0f, a1 = 0.0f;
  for (int i = lane; i < DD; i += 32) {
    float qv = (float)qp[i];
    a0 += qv * (float)k0[i];
    a1 += qv * (float)k1[i];
  }
#pragma unroll
  for (int o = 16; o > 0; o >>= 1) {
    a0 += __shfl_xor(a0, o, 32);
    a1 += __shfl_xor(a1, o, 32);
  }
  const float sc = 0.04419417382415922f;   // 1/sqrt(512)
  a0 *= sc; a1 *= sc;
  float mx = fmaxf(a0, a1);
  float e0 = expf(a0 - mx), e1 = expf(a1 - mx);
  float inv = 1.0f / (e0 + e1);
  float w0 = e0 * inv, w1 = e1 * inv;
  const bf16* v0 = vB + (size_t)(2 * e) * DD;
  const bf16* v1 = v0 + DD;
  float* op = eatt + (size_t)e * DD;
  for (int i = lane; i < DD; i += 32)
    op[i] = w0 * (float)v0[i] + w1 * (float)v1[i];
}

// ------------------------------- batch norm --------------------------------
__global__ __launch_bounds__(256)
void k_bn_partial(const float* __restrict__ X, float* csum, float* csq) {
  int r0 = blockIdx.x * 64;
  int ca = threadIdx.x, cb = threadIdx.x + 256;
  float s1a = 0, s2a = 0, s1b = 0, s2b = 0;
  for (int i = 0; i < 64; ++i) {
    const float* rp = X + (size_t)(r0 + i) * DD;
    float xa = rp[ca], xb = rp[cb];
    s1a += xa; s2a += xa * xa; s1b += xb; s2b += xb * xb;
  }
  atomicAdd(&csum[ca], s1a); atomicAdd(&csq[ca], s2a);
  atomicAdd(&csum[cb], s1b); atomicAdd(&csq[cb], s2b);
}

__global__ void k_bn_norm(const float* __restrict__ X, const float* __restrict__ csum,
                          const float* __restrict__ csq, const float* __restrict__ g,
                          const float* __restrict__ be, bf16* __restrict__ outB, int rows) {
  size_t idx = (size_t)blockIdx.x * 256 + threadIdx.x;
  if (idx >= (size_t)rows * DD) return;
  int c = (int)(idx & (DD - 1));
  float mu = csum[c] / (float)rows;
  float var = csq[c] / (float)rows - mu * mu;
  float y = (X[idx] - mu) * rsqrtf(var + 1e-5f) * g[c] + be[c];
  outB[idx] = (bf16)y;
}

// ===========================================================================
extern "C" void kernel_launch(void* const* d_in, const int* in_sizes, int n_in,
                              void* d_out, int out_size, void* d_ws, size_t ws_size,
                              hipStream_t stream)
{
  (void)in_sizes; (void)n_in; (void)out_size; (void)ws_size;
  const float* feats_in = (const float*)d_in[0];
  const int*   eidx     = (const int*)d_in[1];
  const float* eattr_in = (const float*)d_in[2];
  float* outF = (float*)d_out;

  // Param leaves: JAX pytree flatten order (dict keys sorted) starting at 3:
  // 3:bn1.g 4:bn1.b 5:bn2.g 6:bn2.b 7:cls.W 8:cls.b 9:dim_red.W 10:dim_red.b
  // 11:fc.W 12:fc.b, then per layer l (lb = 13 + 32*l):
  //   edge: ak(0,1) aq(2,3) av(4,5) f1(6,7) f2(8,9) fln(10,11) ln(12,13) n2e(14,15)
  //   node: e2n(16,17) k(18,19) l1(20,21) l2(22,23) ln(24,25) o(26,27) q(28,29) v(30,31)
  auto P = [&](int i) { return (const float*)d_in[i]; };

  size_t cur = 0;
  auto alloc = [&](size_t bytes) -> void* {
    cur = (cur + 255) & ~(size_t)255;
    void* p = (char*)d_ws + cur;
    cur += bytes;
    return p;
  };

  int*  rA = (int*)alloc((size_t)EE * 4);
  int*  cA = (int*)alloc((size_t)EE * 4);
  bf16* featsInB = (bf16*)alloc((size_t)NN * DIN * 2);
  float* featsF = (float*)alloc((size_t)NN * DD * 4);
  bf16*  featsB = (bf16*)alloc((size_t)NN * DD * 2);
  float* edgeF  = (float*)alloc((size_t)EE * DD * 4);
  bf16*  edgeB  = (bf16*)alloc((size_t)EE * DD * 2);

  bf16* dimredT = (bf16*)alloc((size_t)DIN * DD * 2);
  bf16* fcT  = (bf16*)alloc((size_t)DD * DD * 2);
  bf16* clsT = (bf16*)alloc((size_t)NCLSP * DD * 2);   // padded rows zeroed
  struct LWT { bf16 *akT,*aqT,*avT,*f1T,*f2T,*n2eT,*e2nT,*kT,*l1T,*l2T,*oT,*qT,*vT; } lw[2];
  for (int l = 0; l < 2; ++l) {
    lw[l].akT  = (bf16*)alloc((size_t)DD * DD * 2);
    lw[l].aqT  = (bf16*)alloc((size_t)DD * DD * 2);
    lw[l].avT  = (bf16*)alloc((size_t)DD * DD * 2);
    lw[l].f1T  = (bf16*)alloc((size_t)DD * DHID * 2);
    lw[l].f2T  = (bf16*)alloc((size_t)DHID * DD * 2);
    lw[l].n2eT = (bf16*)alloc((size_t)DD * DD * 2);
    lw[l].e2nT = (bf16*)alloc((size_t)DD * DD * 2);
    lw[l].kT   = (bf16*)alloc((size_t)DD * DD * 2);
    lw[l].l1T  = (bf16*)alloc((size_t)DD * DHID * 2);
    lw[l].l2T  = (bf16*)alloc((size_t)DHID * DD * 2);
    lw[l].oT   = (bf16*)alloc((size_t)DD * DD * 2);
    lw[l].qT   = (bf16*)alloc((size_t)DD * DD * 2);
    lw[l].vT   = (bf16*)alloc((size_t)DD * DD * 2);
  }

  bf16* qB = (bf16*)alloc((size_t)NN * DD * 2);
  bf16* kB = (bf16*)alloc((size_t)NN * DD * 2);
  bf16* vB = (bf16*)alloc((size_t)NN * DD * 2);
  float* sE   = (float*)alloc((size_t)EE * HH * 4);
  float* eExp = (float*)alloc((size_t)EE * HH * 4);
  unsigned* mEnc = (unsigned*)alloc((size_t)NN * HH * 4);
  float* mDec = (float*)alloc((size_t)NN * HH * 4);
  float* zSum = (float*)alloc((size_t)NN * HH * 4);
  float* aggF = (float*)alloc((size_t)NN * DD * 4);
  bf16*  aggB = (bf16*)alloc((size_t)NN * DD * 2);
  float* attF = (float*)alloc((size_t)NN * DD * 4);
  float* f1F  = (float*)alloc((size_t)NN * DD * 4);
  float* segF = (float*)alloc((size_t)NN * DD * 4);
  float* fattF = (float*)alloc((size_t)NN * DD * 4);
  bf16*  fattB = (bf16*)alloc((size_t)NN * DD * 2);
  bf16*  hidNB = (bf16*)alloc((size_t)NN * DHID * 2);
  float* ffF  = (float*)alloc((size_t)NN * DD * 4);
  bf16*  efB  = (bf16*)alloc((size_t)NN * DD * 2);

  int*  kvIdx = (int*)alloc((size_t)2 * CEC * 4);
  bf16* qcB   = (bf16*)alloc((size_t)CEC * DD * 2);
  bf16* kcB   = (bf16*)alloc((size_t)2 * CEC * DD * 2);
  bf16* vcB   = (bf16*)alloc((size_t)2 * CEC * DD * 2);
  float* eatF = (float*)alloc((size_t)CEC * DD * 4);
  float* hidF = (float*)alloc((size_t)CEC * DHID * 4);
  bf16*  hidB = (bf16*)alloc((size_t)CEC * DHID * 2);
  float* houtF = (float*)alloc((size_t)CEC * DD * 4);
  float* bnBuf = (float*)alloc((size_t)4 * DD * 4);
  float* bnSumE = bnBuf;            float* bnSqE = bnBuf + DD;
  float* bnSumN = bnBuf + 2 * DD;   float* bnSqN = bnBuf + 3 * DD;

  auto gemm = [&](const bf16* A, const bf16* BT, const float* bias,
                  const int* aidx, const int* sidx,
                  float* Cf, bf16* Cb, int M, int Ncol, int K, int act) {
    dim3 g((unsigned)(M / 128), (unsigned)((Ncol + 127) / 128));
    k_gemm<<<g, 256, 0, stream>>>(A, BT, bias, aidx, sidx, Cf, Cb, M, Ncol, K, act);
  };
  auto norm = [&](const float* a, const float* b, const float* g, const float* be,
                  float* oF, bf16* oB, int rows, int R, int doGelu) {
    k_norm_rows<<<dim3((unsigned)rows), 256, 0, stream>>>(a, b, g, be, oF, oB, R, doGelu);
  };
  auto wT = [&](const float* W, bf16* WTp, int K, int Ncol) {
    size_t n = (size_t)K * Ncol;
    k_wT<<<dim3((unsigned)((n + 255) / 256)), 256, 0, stream>>>(W, WTp, K, Ncol);
  };
  auto cvt = [&](const float* x, bf16* y, size_t n) {
    k_cvt<<<dim3((unsigned)((n + 255) / 256)), 256, 0, stream>>>(x, y, n);
  };

  // ---- input prep + weight transposes -------------------------------------
  k_split<<<dim3((EE + 255) / 256), 256, 0, stream>>>(eidx, rA, cA);
  cvt(feats_in, featsInB, (size_t)NN * DIN);
  cvt(eattr_in, edgeB, (size_t)EE * DD);
  hipMemcpyAsync(edgeF, eattr_in, (size_t)EE * DD * 4, hipMemcpyDeviceToDevice, stream);

  wT(P(9), dimredT, DIN, DD);
  wT(P(11), fcT, DD, DD);
  k_zero_bf16<<<dim3((unsigned)(((size_t)NCLSP * DD + 255) / 256)), 256, 0, stream>>>(
      clsT, (size_t)NCLSP * DD);
  wT(P(7), clsT, DD, NCLS);   // fills first 98 rows of [NCLSP][DD]... (row n of WT)
  for (int l = 0; l < 2; ++l) {
    int lb = 13 + 32 * l;
    wT(P(lb + 0),  lw[l].akT,  DD, DD);
    wT(P(lb + 2),  lw[l].aqT,  DD, DD);
    wT(P(lb + 4),  lw[l].avT,  DD, DD);
    wT(P(lb + 6),  lw[l].f1T,  DD, DHID);
    wT(P(lb + 8),  lw[l].f2T,  DHID, DD);
    wT(P(lb + 14), lw[l].n2eT, DD, DD);
    wT(P(lb + 16), lw[l].e2nT, DD, DD);
    wT(P(lb + 18), lw[l].kT,   DD, DD);
    wT(P(lb + 20), lw[l].l1T,  DD, DHID);
    wT(P(lb + 22), lw[l].l2T,  DHID, DD);
    wT(P(lb + 26), lw[l].oT,   DD, DD);
    wT(P(lb + 28), lw[l].qT,   DD, DD);
    wT(P(lb + 30), lw[l].vT,   DD, DD);
  }

  // ---- dim_red ------------------------------------------------------------
  gemm(featsInB, dimredT, P(10), nullptr, nullptr, featsF, featsB, NN, DD, DIN, 0);

  for (int l = 0; l < 2; ++l) {
    const int lb = 13 + 32 * l;

    // ================= node layer =================
    gemm(featsB, lw[l].qT, P(lb + 29), nullptr, nullptr, nullptr, qB, NN, DD, DD, 0);
    gemm(featsB, lw[l].kT, P(lb + 19), nullptr, nullptr, nullptr, kB, NN, DD, DD, 0);
    gemm(featsB, lw[l].vT, P(lb + 31), nullptr, nullptr, nullptr, vB, NN, DD, DD, 0);

    k_sdot<<<dim3((EE * HH + 255) / 256), 256, 0, stream>>>(qB, kB, rA, cA, sE);
    hipMemsetAsync(mEnc, 0, (size_t)NN * HH * 4, stream);
    k_segmax<<<dim3((EE * HH + 255) / 256), 256, 0, stream>>>(sE, cA, mEnc);
    k_mdec<<<dim3((NN * HH + 255) / 256), 256, 0, stream>>>(mEnc, mDec);
    hipMemsetAsync(zSum, 0, (size_t)NN * HH * 4, stream);
    k_expz<<<dim3((EE * HH + 255) / 256), 256, 0, stream>>>(sE, mDec, cA, eExp, zSum);
    hipMemsetAsync(aggF, 0, (size_t)NN * DD * 4, stream);
    k_scatter_v<<<dim3((unsigned)(((size_t)EE * DD + 255) / 256)), 256, 0, stream>>>(
        eExp, zSum, vB, rA, cA, aggF);
    cvt(aggF, aggB, (size_t)NN * DD);

    gemm(aggB, lw[l].oT, P(lb + 27), nullptr, nullptr, attF, nullptr, NN, DD, DD, 0);
    norm(featsF, attF, P(lb + 24), P(lb + 25), f1F, nullptr, NN, DD, 1);

    // na = gelu(edge_attr @ e2n + b) scatter-added by r into segF (fused)
    hipMemsetAsync(segF, 0, (size_t)NN * DD * 4, stream);
    gemm(edgeB, lw[l].e2nT, P(lb + 17), nullptr, rA, segF, nullptr, EE, DD, DD, 1);
    norm(f1F, segF, P(lb + 24), P(lb + 25), fattF, fattB, NN, DD, 1);

    gemm(fattB, lw[l].l1T, P(lb + 21), nullptr, nullptr, nullptr, hidNB, NN, DHID, DD, 1);
    gemm(hidNB, lw[l].l2T, P(lb + 23), nullptr, nullptr, ffF, nullptr, NN, DD, DHID, 0);
    norm(fattF, ffF, P(lb + 24), P(lb + 25), featsF, featsB, NN, DD, 1);

    // ================= edge layer =================
    gemm(featsB, lw[l].n2eT, P(lb + 15), nullptr, nullptr, nullptr, efB, NN, DD, DD, 1);

    for (int ci = 0; ci < EE / CEC; ++ci) {
      const size_t e0 = (size_t)ci * CEC;
      k_kvidx<<<dim3((CEC + 255) / 256), 256, 0, stream>>>(rA + e0, cA + e0, kvIdx, CEC);
      gemm(edgeB + e0 * DD, lw[l].aqT, P(lb + 3), nullptr, nullptr,
           nullptr, qcB, CEC, DD, DD, 0);
      gemm(efB, lw[l].akT, P(lb + 1), kvIdx, nullptr, nullptr, kcB, 2 * CEC, DD, DD, 0);
      gemm(efB, lw[l].avT, P(lb + 5), kvIdx, nullptr, nullptr, vcB, 2 * CEC, DD, DD, 0);
      k_edge_attn<<<dim3(CEC / 8), 256, 0, stream>>>(qcB, kcB, vcB, eatF, CEC);
      // e1 = LN(gelu(eatt + edge_attr))  (in-place into edgeF/edgeB chunk)
      norm(eatF, edgeF + e0 * DD, P(lb + 12), P(lb + 13),
           edgeF + e0 * DD, edgeB + e0 * DD, CEC, DD, 1);
      // FFN: hid = LN_fln(gelu(e1@f1+b)) ; hout = hid@f2+b
      gemm(edgeB + e0 * DD, lw[l].f1T, P(lb + 7), nullptr, nullptr,
           hidF, nullptr, CEC, DHID, DD, 0);
      norm(hidF, nullptr, P(lb + 10), P(lb + 11), nullptr, hidB, CEC, DHID, 1);
      gemm(hidB, lw[l].f2T, P(lb + 9), nullptr, nullptr, houtF, nullptr, CEC, DD, DHID, 0);
      // edge_attr = LN(gelu(hout + e1))
      norm(houtF, edgeF + e0 * DD, P(lb + 12), P(lb + 13),
           edgeF + e0 * DD, edgeB + e0 * DD, CEC, DD, 1);
    }
  }

  // ---- heads: batch-norms + fc(sigmoid) + cls -----------------------------
  hipMemsetAsync(bnBuf, 0, (size_t)4 * DD * 4, stream);
  k_bn_partial<<<dim3(EE / 64), 256, 0, stream>>>(edgeF, bnSumE, bnSqE);
  k_bn_norm<<<dim3((unsigned)(((size_t)EE * DD + 255) / 256)), 256, 0, stream>>>(
      edgeF, bnSumE, bnSqE, P(3), P(4), edgeB, EE);
  k_bn_partial<<<dim3(NN / 64), 256, 0, stream>>>(featsF, bnSumN, bnSqN);
  k_bn_norm<<<dim3((unsigned)(((size_t)NN * DD + 255) / 256)), 256, 0, stream>>>(
      featsF, bnSumN, bnSqN, P(5), P(6), featsB, NN);

  // node_pred = node_f @ cls + b          -> d_out[0 : N*98)
  gemm(featsB, clsT, P(8), nullptr, nullptr, outF, nullptr, NN, NCLS, DD, 0);
  // edge_para = sigmoid(edge_f @ fc + b)  -> d_out[N*98 : N*98 + E*512)
  gemm(edgeB, fcT, P(12), nullptr, nullptr, outF + (size_t)NN * NCLS, nullptr,
       EE, DD, DD, 2);
}
